// SparseAttention_12799002542746
// MI455X (gfx1250) — matrix operand order repaired and checked by reading, not compiled, not run on hardware
//
#include <hip/hip_runtime.h>

#define SEQ    2048
#define DMODEL 1024
#define NHEAD  16
#define HDIM   64
#define BATCH  2
#define MROWS  (BATCH * SEQ)   // 4096

typedef __attribute__((ext_vector_type(16))) __bf16 v16bf;
typedef __attribute__((ext_vector_type(8)))  __bf16 v8bf;
typedef __attribute__((ext_vector_type(8)))  float  v8f;
typedef unsigned int u32x4 __attribute__((ext_vector_type(4)));
typedef int          i32x4 __attribute__((ext_vector_type(4)));
typedef int          i32x8 __attribute__((ext_vector_type(8)));

union AFrag { v16bf v; v8bf h[2]; __bf16 e[16]; };

// native f32 -> bf16 (RTNE); clang pairs these into v_cvt_pk_bf16_f32
static __device__ inline __bf16 f2bf(float f) { return (__bf16)f; }

// ---------------------------------------------------------------------------
// Pack int32 mask [S][S] into bitmask [S][S/32] (bit j of word w = col w*32+j)
// ---------------------------------------------------------------------------
__global__ __launch_bounds__(256)
void pack_mask(const int* __restrict__ mask, unsigned* __restrict__ bits)
{
  const int w = blockIdx.x * 256 + threadIdx.x;      // S*S/32 = 131072 words
  const int* src = mask + (size_t)w * 32;
  unsigned b = 0;
  #pragma unroll
  for (int i = 0; i < 32; ++i) b |= (src[i] != 0 ? 1u : 0u) << i;
  bits[w] = b;
}

// ---------------------------------------------------------------------------
// Y[M,1024] = X[M,1024] @ W[1024,1024] + bias ; OUT_F32 selects f32 vs bf16 out
// block = 256 threads (8 waves = 4M x 2N). Block tile 128x128; wave tile 32x64.
// ---------------------------------------------------------------------------
template <int OUT_F32>
__global__ __launch_bounds__(256)
void proj_gemm(const float* __restrict__ X, const float* __restrict__ W,
               const float* __restrict__ bias, void* __restrict__ Yv)
{
  constexpr int LDA = 40;                            // bf16 elems per LDS row
  __shared__ __align__(16) __bf16 sA[128 * LDA];     // X tile [128 rows][32 k]
  __shared__ __align__(16) __bf16 sB[128 * LDA];     // W tile transposed [128 n][32 k]

  const int tid  = threadIdx.x;
  const int wave = tid >> 5;
  const int lane = tid & 31;
  const int lh   = lane & 15;
  const int hk   = lane >> 4;
  const int wm   = wave & 3;        // wave row (4)
  const int wn   = wave >> 2;       // wave col (2)

  const int m_base = blockIdx.x * 128;
  const int n_base = blockIdx.y * 128;

  v8f acc[2][4];
  #pragma unroll
  for (int ma = 0; ma < 2; ++ma)
    #pragma unroll
    for (int nf = 0; nf < 4; ++nf)
      #pragma unroll
      for (int r = 0; r < 8; ++r) acc[ma][nf][r] = 0.f;

  for (int kt = 0; kt < 1024; kt += 32) {
    // stage X tile (128x32 f32 -> bf16): float4 loads, packed b64 LDS stores
    #pragma unroll
    for (int i = 0; i < 4; ++i) {
      int e = (tid + i * 256) * 4;
      int r = e >> 5, c = e & 31;
      float4 xv = *(const float4*)&X[(size_t)(m_base + r) * 1024 + kt + c];
      union { __bf16 b[4]; unsigned long long u; } pk;
      pk.b[0] = f2bf(xv.x); pk.b[1] = f2bf(xv.y);
      pk.b[2] = f2bf(xv.z); pk.b[3] = f2bf(xv.w);
      *(unsigned long long*)&sA[r * LDA + c] = pk.u;
    }
    // stage W tile (32x128): float4 loads, transposed scalar stores sB[n][k]
    #pragma unroll
    for (int i = 0; i < 4; ++i) {
      int e = (tid + i * 256) * 4;
      int kk = e >> 7, nn = e & 127;
      float4 wv = *(const float4*)&W[(size_t)(kt + kk) * 1024 + n_base + nn];
      sB[(nn + 0) * LDA + kk] = f2bf(wv.x);
      sB[(nn + 1) * LDA + kk] = f2bf(wv.y);
      sB[(nn + 2) * LDA + kk] = f2bf(wv.z);
      sB[(nn + 3) * LDA + kk] = f2bf(wv.w);
    }
    __syncthreads();

    // A fragments: lane lh = row, halves: K = 8*hk+[0..7] then 16+8*hk+[0..7]
    AFrag a[2];
    #pragma unroll
    for (int ma = 0; ma < 2; ++ma) {
      const int arow = wm * 32 + ma * 16 + lh;
      a[ma].h[0] = *(const v8bf*)&sA[arow * LDA + 8 * hk];
      a[ma].h[1] = *(const v8bf*)&sA[arow * LDA + 16 + 8 * hk];
    }
    #pragma unroll
    for (int nf = 0; nf < 4; ++nf) {
      // B fragment: lane lh = col N, halves: K = 8*hk+[0..7] then 16+8*hk+[0..7]
      AFrag b;
      const int brow = wn * 64 + nf * 16 + lh;
      b.h[0] = *(const v8bf*)&sB[brow * LDA + 8 * hk];
      b.h[1] = *(const v8bf*)&sB[brow * LDA + 16 + 8 * hk];
      #pragma unroll
      for (int ma = 0; ma < 2; ++ma)
        acc[ma][nf] = __builtin_amdgcn_wmma_f32_16x16x32_bf16(
            false, a[ma].v, false, b.v, (short)0, acc[ma][nf], false, false);
    }
    __syncthreads();
  }

  // epilogue: C layout -> lane lh = col, VGPR r -> row r + 8*hk
  #pragma unroll
  for (int ma = 0; ma < 2; ++ma)
    #pragma unroll
    for (int nf = 0; nf < 4; ++nf) {
      int col = n_base + wn * 64 + nf * 16 + lh;
      float bv = bias[col];
      #pragma unroll
      for (int r = 0; r < 8; ++r) {
        int row = m_base + wm * 32 + ma * 16 + r + 8 * hk;
        float val = acc[ma][nf][r] + bv;
        if (OUT_F32) ((float*)Yv)[(size_t)row * 1024 + col] = val;
        else         ((__bf16*)Yv)[(size_t)row * 1024 + col] = f2bf(val);
      }
    }
}

// ---------------------------------------------------------------------------
// Masked flash attention: one wave32 per (b,h, 16-query tile).
// V tile DMA'd to LDS by the Tensor Data Mover; mask read from bitmask.
// ---------------------------------------------------------------------------
__global__ __launch_bounds__(32)
void sparse_flash_attn(const __bf16* __restrict__ qp,
                       const __bf16* __restrict__ kp,
                       const __bf16* __restrict__ vp,
                       const unsigned* __restrict__ mbits,
                       float*        __restrict__ ctx)
{
  constexpr int LP = 40;
  __shared__ __align__(16) __bf16 Pt[16 * LP];       // P tile [16 rows][32 keys]
  __shared__ __align__(16) __bf16 Vt[32 * HDIM];     // V tile row-major [32 keys][64 dims]

  const int lane = threadIdx.x;
  const int lh   = lane & 15;
  const int hk   = lane >> 4;

  const int qtile = blockIdx.x & (SEQ / 16 - 1);     // 128 query tiles
  const int bh    = blockIdx.x >> 7;                 // 0..31
  const int b     = bh >> 4;
  const int h     = bh & (NHEAD - 1);

  const int q0 = qtile * 16;
  const __bf16* qbase = qp + (size_t)(b * SEQ) * DMODEL + h * HDIM;
  const __bf16* kbase = kp + (size_t)(b * SEQ) * DMODEL + h * HDIM;
  const __bf16* vbase = vp + (size_t)(b * SEQ) * DMODEL + h * HDIM;

  // Q fragments for head-dim halves [0,32) and [32,64)
  AFrag qa[2];
  {
    const __bf16* qr = qbase + (size_t)(q0 + lh) * DMODEL;
    #pragma unroll
    for (int c = 0; c < 2; ++c) {
      qa[c].h[0] = *(const v8bf*)(qr + c * 32 + 8 * hk);
      qa[c].h[1] = *(const v8bf*)(qr + c * 32 + 16 + 8 * hk);
    }
  }

  // Tensor DMA descriptor pieces that don't change per iteration
  const unsigned lds_vt = (unsigned)(size_t)(&Vt[0]);   // low 32 bits = LDS offset

  float mrow[8], lrow[8];
  v8f oacc[4];
  #pragma unroll
  for (int r = 0; r < 8; ++r) { mrow[r] = -3.0e38f; lrow[r] = 0.f; }
  #pragma unroll
  for (int nf = 0; nf < 4; ++nf)
    #pragma unroll
    for (int r = 0; r < 8; ++r) oacc[nf][r] = 0.f;

  for (int t = 0; t < SEQ; t += 32) {
    // ---- TDM: DMA V tile (32 keys x 64 dims bf16) from global to LDS ----
    {
      unsigned long long ga =
          (unsigned long long)(size_t)(vbase + (size_t)t * DMODEL);
      u32x4 g0;
      g0[0] = 1u;                                   // count=1 (valid user D#)
      g0[1] = lds_vt;                               // lds_addr
      g0[2] = (unsigned)(ga & 0xFFFFFFFFu);         // global_addr[31:0]
      g0[3] = (unsigned)((ga >> 32) & 0x01FFFFFFu)  // global_addr[56:32]
              | (2u << 30);                         // type = 2 ("image")
      i32x8 g1;
      g1[0] = (int)(1u << 16);                      // data_size = 2 bytes
      g1[1] = (int)(64u << 16);                     // tensor_dim0 = 64 (bits 79:48 lo)
      g1[2] = (int)(32u << 16);                     // tensor_dim1 = 32 (bits 111:80 lo)
      g1[3] = (int)(64u << 16);                     // tile_dim0 = 64 (bits 127:112)
      g1[4] = 32;                                   // tile_dim1 = 32 (bits 143:128)
      g1[5] = 1024;                                 // tensor_dim0_stride = D
      g1[6] = 0;
      g1[7] = 0;
      i32x4 g2; g2[0] = 0; g2[1] = 0; g2[2] = 0; g2[3] = 0;
      i32x4 g3; g3[0] = 0; g3[1] = 0; g3[2] = 0; g3[3] = 0;
      i32x8 g4;
      #pragma unroll
      for (int i = 0; i < 8; ++i) g4[i] = 0;
      __builtin_amdgcn_tensor_load_to_lds(g0, g1, g2, g3, g4, 0);
    }
    if (t + 32 < SEQ) {   // CDNA5 global_prefetch_b8 of next K tile
      __builtin_prefetch(kbase + (size_t)(t + 32 + lane) * DMODEL, 0, 1);
    }

    // scores for the two 16-key subtiles (2 WMMAs each over head dim 64)
    v8f sc[2];
    #pragma unroll
    for (int sub = 0; sub < 2; ++sub) {
      const __bf16* kr = kbase + (size_t)(t + sub * 16 + lh) * DMODEL;
      v8f s;
      #pragma unroll
      for (int r = 0; r < 8; ++r) s[r] = 0.f;
      #pragma unroll
      for (int c = 0; c < 2; ++c) {
        AFrag kb;
        kb.h[0] = *(const v8bf*)(kr + c * 32 + 8 * hk);
        kb.h[1] = *(const v8bf*)(kr + c * 32 + 16 + 8 * hk);
        s = __builtin_amdgcn_wmma_f32_16x16x32_bf16(
            false, qa[c].v, false, kb.v, (short)0, s, false, false);
      }
      sc[sub] = s;
    }

    // mask (bitmask) + scale + online softmax update across the 32 keys
    float p[2][8];
    #pragma unroll
    for (int r = 0; r < 8; ++r) {
      const int row = q0 + r + 8 * hk;
      const unsigned mw = mbits[(size_t)row * (SEQ / 32) + (t >> 5)];
      const bool m0 = (mw >> lh) & 1u;
      const bool m1 = (mw >> (16 + lh)) & 1u;
      float f0 = sc[0][r] * 0.125f;                 // 1/sqrt(64)
      float f1 = sc[1][r] * 0.125f;
      f0 = m0 ? f0 : -3.0e38f;
      f1 = m1 ? f1 : -3.0e38f;
      float tm = fmaxf(f0, f1);
      #pragma unroll
      for (int ofs = 1; ofs < 16; ofs <<= 1)
        tm = fmaxf(tm, __shfl_xor(tm, ofs, 16));
      const float newm  = fmaxf(mrow[r], tm);
      const float alpha = __expf(mrow[r] - newm);
      const float p0 = m0 ? __expf(f0 - newm) : 0.f;
      const float p1 = m1 ? __expf(f1 - newm) : 0.f;
      float rs = p0 + p1;
      #pragma unroll
      for (int ofs = 1; ofs < 16; ofs <<= 1)
        rs += __shfl_xor(rs, ofs, 16);
      lrow[r] = lrow[r] * alpha + rs;
      mrow[r] = newm;
      #pragma unroll
      for (int nf = 0; nf < 4; ++nf) oacc[nf][r] *= alpha;
      p[0][r] = p0; p[1][r] = p1;
    }

    // write P (C layout) into LDS to re-read in A layout
    #pragma unroll
    for (int r = 0; r < 8; ++r) {
      const int prow = r + 8 * hk;
      Pt[prow * LP + lh]      = f2bf(p[0][r]);
      Pt[prow * LP + 16 + lh] = f2bf(p[1][r]);
    }
    __syncthreads();
    __builtin_amdgcn_s_wait_tensorcnt(0);           // V tile resident in LDS

    // O += P(16x32) @ V(32x64)
    AFrag pa;
    pa.h[0] = *(const v8bf*)&Pt[lh * LP + 8 * hk];
    pa.h[1] = *(const v8bf*)&Pt[lh * LP + 16 + 8 * hk];
    #pragma unroll
    for (int nf = 0; nf < 4; ++nf) {
      AFrag vb;   // B frag: lane lh = dim col, elem e: key = 8*hk + e (+8 if e >= 8)
      #pragma unroll
      for (int e = 0; e < 16; ++e)
        vb.e[e] = Vt[(8 * hk + e + (e >= 8 ? 8 : 0)) * HDIM + nf * 16 + lh];
      oacc[nf] = __builtin_amdgcn_wmma_f32_16x16x32_bf16(
          false, pa.v, false, vb.v, (short)0, oacc[nf], false, false);
    }
    __syncthreads();
  }

  // normalize and write context [B,S,D] at this head's 64-dim slice
  float* cb = ctx + (size_t)(b * SEQ) * DMODEL + h * HDIM;
  #pragma unroll
  for (int nf = 0; nf < 4; ++nf) {
    #pragma unroll
    for (int r = 0; r < 8; ++r) {
      const int row = q0 + r + 8 * hk;
      cb[(size_t)row * DMODEL + nf * 16 + lh] = oacc[nf][r] / lrow[r];
    }
  }
}

// ---------------------------------------------------------------------------
extern "C" void kernel_launch(void* const* d_in, const int* in_sizes, int n_in,
                              void* d_out, int out_size, void* d_ws, size_t ws_size,
                              hipStream_t stream) {
  (void)in_sizes; (void)n_in; (void)out_size; (void)ws_size;
  const float* query = (const float*)d_in[0];
  const float* key_  = (const float*)d_in[1];
  const float* value = (const float*)d_in[2];
  const int*   mask  = (const int*)d_in[3];
  const float* Wq = (const float*)d_in[4];
  const float* bq = (const float*)d_in[5];
  const float* Wk = (const float*)d_in[6];
  const float* bk = (const float*)d_in[7];
  const float* Wv = (const float*)d_in[8];
  const float* bv = (const float*)d_in[9];
  const float* Wo = (const float*)d_in[10];
  const float* bo = (const float*)d_in[11];

  char* ws = (char*)d_ws;
  __bf16*   qp    = (__bf16*)(ws);                           // 8 MB
  __bf16*   kp    = (__bf16*)(ws + 8ull  * 1024 * 1024);     // 8 MB
  __bf16*   vp    = (__bf16*)(ws + 16ull * 1024 * 1024);     // 8 MB
  float*    ctx   = (float*) (ws + 24ull * 1024 * 1024);     // 16 MB
  unsigned* mbits = (unsigned*)(ws + 40ull * 1024 * 1024);   // 512 KB

  pack_mask<<<(SEQ * SEQ / 32) / 256, 256, 0, stream>>>(mask, mbits);

  dim3 gg(MROWS / 128, 1024 / 128);   // (32, 8)
  proj_gemm<0><<<gg, 256, 0, stream>>>(query, Wq, bq, (void*)qp);
  proj_gemm<0><<<gg, 256, 0, stream>>>(key_,  Wk, bk, (void*)kp);
  proj_gemm<0><<<gg, 256, 0, stream>>>(value, Wv, bv, (void*)vp);

  sparse_flash_attn<<<BATCH * NHEAD * (SEQ / 16), 32, 0, stream>>>(
      qp, kp, vp, mbits, ctx);

  proj_gemm<1><<<gg, 256, 0, stream>>>(ctx, Wo, bo, d_out);
}
